// Gemma4AudioAttention_83562883711248
// MI455X (gfx1250) — matrix operand order repaired and checked by reading, compiled, not run on hardware
//
#include <hip/hip_runtime.h>
#include <stdint.h>

// ---------------- problem constants ----------------
#define BATCH   8
#define TSEQ    2048
#define HIDDIM  1536
#define NHEAD   8
#define HEADD   192
#define NCHUNK  12
#define NPAST   12
#define NCTX    24      // CHUNK + PAST + FUT
#define NF      13      // PAST + FUT + 1
#define NU      171     // ceil(TSEQ / CHUNK)
#define N3      4608    // 3 * HID
#define MROWS   (BATCH * TSEQ)   // 16384
#define NKT     48      // 1536 / 32 K-tiles
#define SOFTCAP 50.0f
#define INVALID (-1.0e9f)

typedef __bf16 v16bf __attribute__((ext_vector_type(16)));
typedef float  v8f   __attribute__((ext_vector_type(8)));

union FragBF { v16bf v; unsigned int u[8]; };
union FragF  { v8f   v; float        f[8]; };

// CDNA5 async global->LDS copy (ASYNCcnt-tracked), guarded so we compile on
// toolchains without the gfx1250 async builtins.
#if defined(__has_builtin)
#if __has_builtin(__builtin_amdgcn_global_load_async_to_lds_b128) && \
    __has_builtin(__builtin_amdgcn_s_wait_asynccnt)
#define USE_ASYNC_COPY 1
#endif
#endif
#ifndef USE_ASYNC_COPY
#define USE_ASYNC_COPY 0
#endif

// exact parameter types per hipcc diagnostic: 16-byte int vector pointers,
// global (AS1) source and LDS (AS3) destination
typedef int v4i_t __attribute__((vector_size(16)));
typedef __attribute__((address_space(1))) v4i_t g_v4i;
typedef __attribute__((address_space(3))) v4i_t l_v4i;

// ---------------- f32 -> bf16 (RNE) ----------------
__global__ __launch_bounds__(256) void cvt_bf16_kernel(const float* __restrict__ in,
                                                       unsigned short* __restrict__ out,
                                                       int n) {
  int i = blockIdx.x * 256 + threadIdx.x;
  if (i < n) {
    unsigned int u = __float_as_uint(in[i]);
    unsigned int r = u + 0x7FFFu + ((u >> 16) & 1u);
    out[i] = (unsigned short)(r >> 16);
  }
}

// ---------------- per-dim query scale: D^-0.5/ln2 * softplus(ps) ----------------
__global__ void qscale_kernel(const float* __restrict__ ps, float* __restrict__ qs) {
  int d = threadIdx.x;
  if (d < HEADD) {
    float base = (1.0f / sqrtf((float)HEADD)) / logf(2.0f);
    qs[d] = base * log1pf(expf(ps[d]));
  }
}

// ---------------- sin_emb = (TransformerXL sinusoid) @ W_pos : [13][1536] ----------------
__global__ __launch_bounds__(256) void sinemb_kernel(const float* __restrict__ Wpos,
                                                     float* __restrict__ sinW) {
  __shared__ float sem[HIDDIM];
  int f = blockIdx.x / 6;                 // 13 relative positions
  int o = (blockIdx.x % 6) * 256 + threadIdx.x;
  float pos = (float)(NPAST - f);         // 12,11,...,0
  float kfac = logf(10000.0f) / 767.0f;   // nts-1 = 767
  for (int i = threadIdx.x; i < HIDDIM; i += 256) {
    int j = (i < 768) ? i : i - 768;
    float st = pos * expf(-(float)j * kfac);
    sem[i] = (i < 768) ? sinf(st) : cosf(st);
  }
  __syncthreads();
  float acc = 0.f;
  for (int i = 0; i < HIDDIM; ++i)
    acc += sem[i] * Wpos[(size_t)i * HIDDIM + o];
  sinW[(size_t)f * HIDDIM + o] = acc;
}

// ---------------- GEMM staging helpers ----------------
// A tile: 64 rows x 32 bf16 (16 u32), LDS row pitch 20 u32 (16B-aligned chunks,
// 20*r mod 64 spans 16 distinct banks for fragment reads).
__device__ __forceinline__ void stage_a_issue(const unsigned int* __restrict__ xbf32,
                                              unsigned int* sAbuf, int rowBase, int k0,
                                              int r, int c4, unsigned int* ra) {
  const unsigned int* gp = &xbf32[(size_t)(rowBase + r) * 768 + (k0 >> 1) + c4 * 4];
#if USE_ASYNC_COPY
  __builtin_amdgcn_global_load_async_to_lds_b128((g_v4i*)gp,
                                                 (l_v4i*)&sAbuf[r * 20 + c4 * 4], 0, 0);
  (void)ra;
#else
#pragma unroll
  for (int i = 0; i < 4; ++i) ra[i] = gp[i];
#endif
}

__device__ __forceinline__ void stage_a_commit(unsigned int* sAbuf, int r, int c4,
                                               const unsigned int* ra) {
#if !USE_ASYNC_COPY
#pragma unroll
  for (int i = 0; i < 4; ++i) sAbuf[r * 20 + c4 * 4 + i] = ra[i];
#else
  (void)sAbuf; (void)r; (void)c4; (void)ra;
#endif
}

// B tile: 32 k x 64 n staged *transposed* into LDS ([n][kk], u16, pitch 34).
// Register-staged so the 8 coalesced global loads batch ahead of the stores.
__device__ __forceinline__ void stage_b_load(const unsigned short* __restrict__ wbf16,
                                             int colBase, int k0, int n, int kkb,
                                             unsigned short* rb) {
#pragma unroll
  for (int jj = 0; jj < 8; ++jj)
    rb[jj] = wbf16[(size_t)(k0 + kkb + 4 * jj) * N3 + colBase + n];
}

__device__ __forceinline__ void stage_b_store(unsigned short* sB16buf, int n, int kkb,
                                              const unsigned short* rb) {
#pragma unroll
  for (int jj = 0; jj < 8; ++jj)
    sB16buf[n * 34 + kkb + 4 * jj] = rb[jj];
}

// ---------------- QKV GEMM: [16384,1536]bf16 x [1536,4608]bf16 -> scaled f32 q/k/v ----------------
__global__ __launch_bounds__(256) void qkv_gemm_kernel(const unsigned int* __restrict__ xbf32,   // [M][768] u32-pairs
                                                       const unsigned short* __restrict__ wbf16, // [1536][4608]
                                                       const float* __restrict__ qscale_pd,
                                                       float* __restrict__ qf,
                                                       float* __restrict__ kf,
                                                       float* __restrict__ vf) {
  __shared__ unsigned int sA[2][64 * 20];   // double-buffered A
  __shared__ unsigned int sB[2][64 * 17];   // double-buffered B (transposed, padded)

  const int tid   = threadIdx.x;
  const int lane  = tid & 31;
  const int wid   = tid >> 5;
  const int waveM = wid >> 1;            // 0..3 -> 16-row slabs
  const int waveN = wid & 1;             // 0..1 -> 32-col slabs
  const int hh    = lane >> 4;           // lane half
  const int l15   = lane & 15;
  const int rowBase = blockIdx.x * 64;
  const int colBase = blockIdx.y * 64;

  // per-thread staging coordinates
  const int ar  = tid >> 2;              // A row 0..63
  const int ac4 = tid & 3;               // A 16B chunk 0..3
  const int bn  = tid & 63;              // B column 0..63
  const int bkk = tid >> 6;              // B k base 0..3

  FragF acc0, acc1;
  acc0.v = (v8f){0.f, 0.f, 0.f, 0.f, 0.f, 0.f, 0.f, 0.f};
  acc1.v = acc0.v;

  // ---- prologue: stage tile 0 into buffer 0 ----
  {
    unsigned int   ra[4];
    unsigned short rb[8];
    stage_a_issue(xbf32, sA[0], rowBase, 0, ar, ac4, ra);
    stage_b_load(wbf16, colBase, 0, bn, bkk, rb);
    stage_a_commit(sA[0], ar, ac4, ra);
    stage_b_store((unsigned short*)sB[0], bn, bkk, rb);
#if USE_ASYNC_COPY
    __builtin_amdgcn_s_wait_asynccnt(0);
#endif
    __syncthreads();
  }

  for (int kt = 0; kt < NKT; ++kt) {
    const int cur  = kt & 1;
    const int nxt  = cur ^ 1;
    const bool more = (kt + 1) < NKT;

    unsigned int   ra[4];
    unsigned short rb[8];
    if (more) {
      const int k1 = (kt + 1) * 32;
      stage_a_issue(xbf32, sA[nxt], rowBase, k1, ar, ac4, ra);
      stage_b_load(wbf16, colBase, k1, bn, bkk, rb);
      if (kt + 2 < NKT) {  // global_prefetch_b8 for the tile after next
        __builtin_prefetch(&wbf16[(size_t)((kt + 2) * 32 + bkk) * N3 + colBase + bn], 0, 1);
        __builtin_prefetch(&xbf32[(size_t)(rowBase + ar) * 768 + ((kt + 2) * 16)], 0, 1);
      }
    }

    // ---- fragments from current buffer ----
    // A: lane half 0 holds K 0-7 & 16-23, half 1 holds K 8-15 & 24-31 (ISA 7.12.2)
    FragBF a, b0, b1;
    const int arow = (waveM * 16 + l15) * 20;
#pragma unroll
    for (int j = 0; j < 4; ++j) {
      a.u[j]     = sA[cur][arow + 4 * hh + j];
      a.u[4 + j] = sA[cur][arow + 8 + 4 * hh + j];
    }
    // B: lane = column, half 0 K 0-7 & 16-23, half 1 K 8-15 & 24-31
    const int c0 = (waveN * 32 + l15) * 17;
    const int c1 = (waveN * 32 + 16 + l15) * 17;
#pragma unroll
    for (int j = 0; j < 4; ++j) {
      b0.u[j]     = sB[cur][c0 + 4 * hh + j];
      b0.u[4 + j] = sB[cur][c0 + 8 + 4 * hh + j];
      b1.u[j]     = sB[cur][c1 + 4 * hh + j];
      b1.u[4 + j] = sB[cur][c1 + 8 + 4 * hh + j];
    }

    acc0.v = __builtin_amdgcn_wmma_f32_16x16x32_bf16(false, a.v, false, b0.v,
                                                     (short)0, acc0.v, false, false);
    acc1.v = __builtin_amdgcn_wmma_f32_16x16x32_bf16(false, a.v, false, b1.v,
                                                     (short)0, acc1.v, false, false);

    if (more) {
      stage_a_commit(sA[nxt], ar, ac4, ra);
      stage_b_store((unsigned short*)sB[nxt], bn, bkk, rb);
#if USE_ASYNC_COPY
      __builtin_amdgcn_s_wait_asynccnt(0);
#endif
    }
    __syncthreads();
  }

  // ---- epilogue: apply q/k scales, scatter into q/k/v f32 buffers ----
  const float KSCALE = 1.89460015f;  // log(1+e)/log(2)
#pragma unroll
  for (int half = 0; half < 2; ++half) {
    const FragF& acc = half ? acc1 : acc0;
    int col   = colBase + waveN * 32 + half * 16 + l15;
    int which = col / HIDDIM;       // 0=q 1=k 2=v (uniform per fragment: 1536 % 16 == 0)
    int rem   = col % HIDDIM;
    int d     = rem % HEADD;
    float scale = (which == 0) ? qscale_pd[d] : (which == 1 ? KSCALE : 1.0f);
    float* dst  = (which == 0) ? qf : (which == 1 ? kf : vf);
#pragma unroll
    for (int r = 0; r < 8; ++r) {
      int row = rowBase + waveM * 16 + 8 * hh + r;   // C/D layout: VGPR r -> M = 8*hh + r
      dst[(size_t)row * HIDDIM + rem] = acc.f[r] * scale;
    }
  }
}

// ---------------- local attention per (b,h,u) block ----------------
__global__ __launch_bounds__(64) void local_attn_kernel(const float* __restrict__ qf,
                                                        const float* __restrict__ kf,
                                                        const float* __restrict__ vf,
                                                        const float* __restrict__ sinW,
                                                        const unsigned char* __restrict__ mask,
                                                        float* __restrict__ out) {
  __shared__ float qs[NCHUNK * HEADD];
  __shared__ float ks[NCTX * HEADD];
  __shared__ float vs[NCTX * HEADD];
  __shared__ float ses[NF * HEADD];
  __shared__ float lg[NCHUNK * NCTX];
  __shared__ int   kval[NCTX];

  const int tid = threadIdx.x;
  const int u = blockIdx.x, h = blockIdx.y, b = blockIdx.z;

  // stage q (zero-fill past T), k/v with halo (zero-fill OOB)
  for (int i = tid; i < NCHUNK * HEADD; i += 64) {
    int w = i / HEADD, d = i % HEADD;
    int t = u * NCHUNK + w;
    qs[i] = (t < TSEQ) ? qf[((size_t)(b * TSEQ + t)) * HIDDIM + h * HEADD + d] : 0.f;
  }
  for (int i = tid; i < NCTX * HEADD; i += 64) {
    int c = i / HEADD, d = i % HEADD;
    int t = u * NCHUNK - NPAST + c;
    bool inr = (t >= 0) && (t < TSEQ);
    size_t src = ((size_t)(b * TSEQ + (inr ? t : 0))) * HIDDIM + h * HEADD + d;
    ks[i] = inr ? kf[src] : 0.f;
    vs[i] = inr ? vf[src] : 0.f;
  }
  for (int i = tid; i < NF * HEADD; i += 64) {
    int f = i / HEADD, d = i % HEADD;
    ses[i] = sinW[(size_t)f * HIDDIM + h * HEADD + d];
  }
  if (tid < NCTX) {
    int t = u * NCHUNK - NPAST + tid;
    kval[tid] = (t >= 0 && t < TSEQ && mask[b * TSEQ + t] == 0) ? 1 : 0;
  }
  __syncthreads();

  // logits: content-content + rel-shifted content-position, softcap, mask
  for (int p = tid; p < NCHUNK * NCTX; p += 64) {
    int w = p / NCTX, c = p % NCTX;
    float ac = 0.f;
#pragma unroll 8
    for (int d = 0; d < HEADD; ++d) ac += qs[w * HEADD + d] * ks[c * HEADD + d];
    int f = c - w;                    // rel-shift: tb[w][c] = term_bd[w][c-w] iff 0<=c-w<F
    float bd = 0.f;
    if (f >= 0 && f < NF) {
#pragma unroll 8
      for (int d = 0; d < HEADD; ++d) bd += qs[w * HEADD + d] * ses[f * HEADD + d];
    }
    float lo = ac + bd;
    lo = tanhf(lo / SOFTCAP) * SOFTCAP;
    bool causal = (c >= w) && (c <= w + NPAST);
    lg[p] = (causal && kval[c]) ? lo : INVALID;
  }
  __syncthreads();

  // row softmax over 24 context slots
  if (tid < NCHUNK) {
    float m = INVALID;
    for (int c = 0; c < NCTX; ++c) m = fmaxf(m, lg[tid * NCTX + c]);
    float s = 0.f;
    for (int c = 0; c < NCTX; ++c) {
      float e = __expf(lg[tid * NCTX + c] - m);
      lg[tid * NCTX + c] = e;
      s += e;
    }
    float inv = 1.0f / s;
    for (int c = 0; c < NCTX; ++c) lg[tid * NCTX + c] *= inv;
  }
  __syncthreads();

  // out[w,d] = sum_c p[w,c] * v[c,d]
  for (int i = tid; i < NCHUNK * HEADD; i += 64) {
    int w = i / HEADD, d = i % HEADD;
    int t = u * NCHUNK + w;
    if (t < TSEQ) {
      float o = 0.f;
#pragma unroll
      for (int c = 0; c < NCTX; ++c) o += lg[w * NCTX + c] * vs[c * HEADD + d];
      out[(((size_t)(b * TSEQ + t)) * NHEAD + h) * HEADD + d] = o;
    }
  }
}

// ---------------- host-side launch ----------------
extern "C" void kernel_launch(void* const* d_in, const int* in_sizes, int n_in,
                              void* d_out, int out_size, void* d_ws, size_t ws_size,
                              hipStream_t stream) {
  const float* x             = (const float*)d_in[0];
  const unsigned char* mask  = (const unsigned char*)d_in[1];   // bool [B,T]
  /* d_in[2] causal_valid_mask recomputed analytically */
  const float* W_qkv         = (const float*)d_in[3];
  const float* W_pos         = (const float*)d_in[4];
  const float* per_dim_scale = (const float*)d_in[5];

  char* ws = (char*)d_ws;
  size_t off = 0;
  auto carve = [&](size_t bytes) -> char* {
    char* p = ws + off;
    off = (off + bytes + 255) & ~(size_t)255;
    return p;
  };
  float* qscale        = (float*)carve((size_t)HEADD * 4);
  float* sinW          = (float*)carve((size_t)NF * HIDDIM * 4);
  unsigned short* xbf  = (unsigned short*)carve((size_t)MROWS * HIDDIM * 2);
  unsigned short* wbf  = (unsigned short*)carve((size_t)HIDDIM * N3 * 2);
  float* qf            = (float*)carve((size_t)MROWS * HIDDIM * 4);
  float* kf            = (float*)carve((size_t)MROWS * HIDDIM * 4);
  float* vf            = (float*)carve((size_t)MROWS * HIDDIM * 4);
  (void)ws_size; (void)in_sizes; (void)n_in; (void)out_size;

  const int nx = MROWS * HIDDIM;       // 25165824
  const int nw = HIDDIM * N3;          // 7077888
  cvt_bf16_kernel<<<(nx + 255) / 256, 256, 0, stream>>>(x, xbf, nx);
  cvt_bf16_kernel<<<(nw + 255) / 256, 256, 0, stream>>>(W_qkv, wbf, nw);
  qscale_kernel<<<1, 192, 0, stream>>>(per_dim_scale, qscale);
  sinemb_kernel<<<NF * (HIDDIM / 256), 256, 0, stream>>>(W_pos, sinW);

  qkv_gemm_kernel<<<dim3(MROWS / 64, N3 / 64), 256, 0, stream>>>(
      (const unsigned int*)xbf, wbf, qscale, qf, kf, vf);

  local_attn_kernel<<<dim3(NU, NHEAD, BATCH), 64, 0, stream>>>(
      qf, kf, vf, sinW, mask, (float*)d_out);
}
